// DSALite_76802605187774
// MI455X (gfx1250) — matrix operand order repaired and checked by reading, not compiled, not run on hardware
//
#include <hip/hip_runtime.h>
#include <hip/hip_bf16.h>
#include <math.h>

// ---------------------------------------------------------------------------
// DSA-Lite attention for gfx1250 (MI455X), wave32 + v_wmma_f32_16x16x32_bf16.
// ---------------------------------------------------------------------------

typedef __attribute__((ext_vector_type(16))) __bf16 v16bf;
typedef __attribute__((ext_vector_type(8)))  __bf16 v8bf;
typedef __attribute__((ext_vector_type(8)))  float  v8f;

#define T_SEQ   2048
#define D_MODEL 1024
#define NHEAD   16
#define HDIM    64
#define NEGC    (-1000000000.0f)
// k = int(2048*0.1) = 204; columns repeat in groups of 8 -> kth largest
// distinct downsampled value index = ceil(204/8) = 26.
#define KTH_DS  26

// A-fragment: two 16-byte chunks per lane, adjacent at every call site (k = 16*h + s).
__device__ __forceinline__ v16bf ld_a(const __bf16* p0, const __bf16* p1) {
  v8bf a = *(const v8bf*)p0;
  v8bf b = *(const v8bf*)p1;
  v16bf r;
#pragma unroll
  for (int i = 0; i < 8; ++i) { r[i] = a[i]; r[i + 8] = b[i]; }
  return r;
}
// B-fragment: 32 contiguous bytes per lane (k = 16*h + s, same order as ld_a).
__device__ __forceinline__ v16bf ld_b(const __bf16* p) { return *(const v16bf*)p; }

__device__ __forceinline__ v8f wmma_bf16(v16bf a, v16bf b, v8f c) {
  return __builtin_amdgcn_wmma_f32_16x16x32_bf16(false, a, false, b, (short)0, c,
                                                 false, false);
}

// ---------------------------------------------------------------------------
// Kernel 1: fp32 -> bf16 conversion of x and the four weights; u_scale.
// ---------------------------------------------------------------------------
__global__ void k_convert(const float* __restrict__ x, const float* __restrict__ U,
                          const float* __restrict__ wq, const float* __restrict__ wk,
                          const float* __restrict__ wv, const float* __restrict__ wo,
                          __bf16* __restrict__ xb, __bf16* __restrict__ wqb,
                          __bf16* __restrict__ wkb, __bf16* __restrict__ wvb,
                          __bf16* __restrict__ wob, float* __restrict__ us) {
  int i = blockIdx.x * blockDim.x + threadIdx.x;
  if (i < T_SEQ * D_MODEL) xb[i] = (__bf16)x[i];
  if (i < D_MODEL * D_MODEL) {
    wqb[i] = (__bf16)wq[i];
    wkb[i] = (__bf16)wk[i];
    wvb[i] = (__bf16)wv[i];
    wob[i] = (__bf16)wo[i];
  }
  if (i < T_SEQ) us[i] = 1.0f + fminf(fmaxf(U[i], 0.0f), 1.0f);
}

// ---------------------------------------------------------------------------
// Kernel 2: QKV projections. One wave = one 16x64 output tile (4 accumulators);
// A-fragment loaded once per k-step and reused across 4 B-fragments. Register
// double-buffering: iteration k+32's loads are issued before the 4 WMMAs that
// consume iteration k's registers, so VMEM overlaps the WMMA pipe (partial
// s_wait_loadcnt instead of draining to 0 per WMMA).
// blockIdx.y selects Q/K/V. Q,K stored [h][t][64] bf16; V stored transposed
// [h][64][t] bf16 (so the attn P*V B-operand reads 32 contiguous bytes/lane).
// ---------------------------------------------------------------------------
__global__ void k_proj(const __bf16* __restrict__ xb, const __bf16* __restrict__ wqb,
                       const __bf16* __restrict__ wkb, const __bf16* __restrict__ wvb,
                       __bf16* __restrict__ Qb, __bf16* __restrict__ Kb,
                       __bf16* __restrict__ Vt) {
  int lane = threadIdx.x & 31, wid = threadIdx.x >> 5;
  int lh = lane & 15, hs = lane >> 4;
  int which = blockIdx.y;
  int tile = blockIdx.x * 8 + wid;          // 2048 tiles = 128 x 16
  int tm = tile >> 4, tn = tile & 15;
  int row0 = tm * 16, col0 = tn * 64;
  const __bf16* W = (which == 0) ? wqb : (which == 1) ? wkb : wvb;
  const __bf16* arow = xb + (size_t)(row0 + lh) * D_MODEL + hs * 16;
  const __bf16* br0 = W + (size_t)(col0 +  0 + lh) * D_MODEL + hs * 16;
  const __bf16* br1 = W + (size_t)(col0 + 16 + lh) * D_MODEL + hs * 16;
  const __bf16* br2 = W + (size_t)(col0 + 32 + lh) * D_MODEL + hs * 16;
  const __bf16* br3 = W + (size_t)(col0 + 48 + lh) * D_MODEL + hs * 16;
  v8f a0 = {}, a1 = {}, a2 = {}, a3 = {};
  v16bf ac  = ld_a(arow, arow + 8);
  v16bf b0c = ld_b(br0), b1c = ld_b(br1), b2c = ld_b(br2), b3c = ld_b(br3);
  for (int k = 32; k < D_MODEL; k += 32) {
    v16bf an  = ld_a(arow + k, arow + k + 8);
    v16bf b0n = ld_b(br0 + k), b1n = ld_b(br1 + k);
    v16bf b2n = ld_b(br2 + k), b3n = ld_b(br3 + k);
    a0 = wmma_bf16(ac, b0c, a0);
    a1 = wmma_bf16(ac, b1c, a1);
    a2 = wmma_bf16(ac, b2c, a2);
    a3 = wmma_bf16(ac, b3c, a3);
    ac = an; b0c = b0n; b1c = b1n; b2c = b2n; b3c = b3n;
  }
  a0 = wmma_bf16(ac, b0c, a0);
  a1 = wmma_bf16(ac, b1c, a1);
  a2 = wmma_bf16(ac, b2c, a2);
  a3 = wmma_bf16(ac, b3c, a3);
  int h = col0 >> 6;                        // col0 is a multiple of 64
#pragma unroll
  for (int r = 0; r < 8; ++r) {
    int t = row0 + r + hs * 8;
    float vals[4] = {a0[r], a1[r], a2[r], a3[r]};
#pragma unroll
    for (int c = 0; c < 4; ++c) {
      int hd = c * 16 + lh;
      __bf16 v = (__bf16)vals[c];
      if (which == 0)      Qb[((size_t)h * T_SEQ + t) * HDIM + hd] = v;
      else if (which == 1) Kb[((size_t)h * T_SEQ + t) * HDIM + hd] = v;
      else                 Vt[((size_t)h * HDIM + hd) * T_SEQ + t] = v;
    }
  }
}

// ---------------------------------------------------------------------------
// Kernel 3: downsampled scores Sd[h][i8][j8] = (Q[8*i8] . K[8*j8]) * HD^-0.5
// 16 heads x 256x256, WMMA tiles of 16x16, K-loop = 64 (2 WMMAs).
// ---------------------------------------------------------------------------
__global__ void k_sd(const __bf16* __restrict__ Qb, const __bf16* __restrict__ Kb,
                     float* __restrict__ Sd) {
  int lane = threadIdx.x & 31, wid = threadIdx.x >> 5;
  int lh = lane & 15, hs = lane >> 4;
  int h = blockIdx.y;
  int tile = blockIdx.x * 8 + wid;          // 256 tiles/head
  int tm = tile >> 4, tn = tile & 15;
  const __bf16* arow = Qb + ((size_t)h * T_SEQ + (size_t)(tm * 16 + lh) * 8) * HDIM;
  const __bf16* brow = Kb + ((size_t)h * T_SEQ + (size_t)(tn * 16 + lh) * 8) * HDIM;
  v8f acc = {};
#pragma unroll
  for (int k = 0; k < HDIM; k += 32) {
    v16bf a = ld_a(arow + k + hs * 16, arow + k + hs * 16 + 8);
    v16bf b = ld_b(brow + k + hs * 16);
    acc = wmma_bf16(a, b, acc);
  }
#pragma unroll
  for (int r = 0; r < 8; ++r) {
    int i8 = tm * 16 + r + hs * 8;
    Sd[((size_t)h * 256 + i8) * 256 + tn * 16 + lh] = acc[r] * 0.125f;
  }
}

// ---------------------------------------------------------------------------
// Kernel 4: exact 26th-largest of each 256-element Sd row (one wave per row)
// via bit-exact binary search on order-preserving uint keys.
// ---------------------------------------------------------------------------
__global__ void k_kth(const float* __restrict__ Sd, float* __restrict__ base26) {
  int lane = threadIdx.x & 31, wid = threadIdx.x >> 5;
  int row = blockIdx.x * 4 + wid;           // 4096 rows = 16 heads * 256
  unsigned key[8];
#pragma unroll
  for (int i = 0; i < 8; ++i) {
    unsigned u = __float_as_uint(Sd[(size_t)row * 256 + lane * 8 + i]);
    key[i] = (u & 0x80000000u) ? ~u : (u | 0x80000000u);
  }
  unsigned lo = 0u, hi = 0xFFFFFFFFu;
  while (lo < hi) {
    unsigned mid = (unsigned)(((unsigned long long)lo + (unsigned long long)hi + 1ull) >> 1);
    int c = 0;
#pragma unroll
    for (int i = 0; i < 8; ++i) c += (key[i] >= mid) ? 1 : 0;
#pragma unroll
    for (int m = 1; m < 32; m <<= 1) c += __shfl_xor(c, m, 32);
    if (c >= KTH_DS) lo = mid; else hi = mid - 1u;
  }
  if (lane == 0) {
    unsigned u = (lo & 0x80000000u) ? (lo ^ 0x80000000u) : ~lo;
    base26[row] = __uint_as_float(u);
  }
}

// ---------------------------------------------------------------------------
// Kernel 5: flash attention with on-the-fly DSA mask. One wave = 16 query
// rows x full causal extent, j-steps of 32 (K-dim of the P*V WMMA).
// C-layout row stats (m,l) live per-lane-half; reduced with shfl_xor 1/2/4/8.
// P tile round-trips through LDS to convert C-layout -> A-layout. The V
// B-fragments are hoisted above the softmax math + LDS barrier so their VMEM
// latency hides behind the exp/shfl section.
// ---------------------------------------------------------------------------
__global__ void __launch_bounds__(128)
k_attn(const __bf16* __restrict__ Qb, const __bf16* __restrict__ Kb,
       const __bf16* __restrict__ Vt, const float* __restrict__ Sd,
       const float* __restrict__ base26, const float* __restrict__ us,
       __bf16* __restrict__ attnA) {
  __shared__ __align__(16) __bf16 Pst[4][16][40];   // 80B rows: aligned b128 reads
  int lane = threadIdx.x & 31, wid = threadIdx.x >> 5;
  int lh = lane & 15, hs = lane >> 4;
  int gidx = blockIdx.x * 4 + wid;          // 2048 waves = 16 heads * 128 i-tiles
  int h = gidx >> 7;
  int ibase = (gidx & 127) * 16;
  int off = hs * 8;
  int i8 = (ibase >> 3) + hs;               // this lane's rows share one Sd row
  float b26 = base26[h * 256 + i8];
  const float* sdrow = Sd + ((size_t)h * 256 + i8) * 256;

  float u10[8]; int rowid[8];
#pragma unroll
  for (int r = 0; r < 8; ++r) { rowid[r] = ibase + r + off; u10[r] = us[rowid[r]] * 10.0f; }

  const __bf16* arow = Qb + ((size_t)h * T_SEQ + ibase + lh) * HDIM;
  v16bf Aq0 = ld_a(arow + hs * 16,      arow + hs * 16 + 8);   // d = 0..31
  v16bf Aq1 = ld_a(arow + 32 + hs * 16, arow + 40 + hs * 16);   // d = 32..63

  const __bf16* vbase0 = Vt + ((size_t)h * HDIM +  0 + lh) * T_SEQ + hs * 16;
  const __bf16* vbase1 = Vt + ((size_t)h * HDIM + 16 + lh) * T_SEQ + hs * 16;
  const __bf16* vbase2 = Vt + ((size_t)h * HDIM + 32 + lh) * T_SEQ + hs * 16;
  const __bf16* vbase3 = Vt + ((size_t)h * HDIM + 48 + lh) * T_SEQ + hs * 16;

  float m[8], l[8];
  v8f O0 = {}, O1 = {}, O2 = {}, O3 = {};
#pragma unroll
  for (int r = 0; r < 8; ++r) { m[r] = -INFINITY; l[r] = 0.0f; }

  int jend = ibase + 16;                    // tiles fully above diagonal add exp(NEG)=0
  for (int jb = 0; jb < jend; jb += 32) {
    float s0[8], s1[8];
#pragma unroll
    for (int st = 0; st < 2; ++st) {
      int jt = jb + st * 16;
      const __bf16* brow = Kb + ((size_t)h * T_SEQ + jt + lh) * HDIM;
      v8f c = {};
      c = wmma_bf16(Aq0, ld_b(brow + hs * 16), c);
      c = wmma_bf16(Aq1, ld_b(brow + 32 + hs * 16), c);
      int jcol = jt + lh;
      float sdv = sdrow[jcol >> 3];
      float* s = st ? s1 : s0;
#pragma unroll
      for (int r = 0; r < 8; ++r) {
        float sig  = 1.0f / (1.0f + __expf(-(sdv - b26) * u10[r]));
        float mval = (jcol <= rowid[r]) ? sig : 0.0f;   // causal * sigmoid mask
        s[r] = c[r] * 0.125f + (1.0f - mval) * NEGC;
      }
    }
    // Prefetch the V B-fragments now: independent of P, and the "memory"
    // clobber on the LDS barrier below would otherwise pin them after it.
    v16bf Bv0 = ld_b(vbase0 + jb);
    v16bf Bv1 = ld_b(vbase1 + jb);
    v16bf Bv2 = ld_b(vbase2 + jb);
    v16bf Bv3 = ld_b(vbase3 + jb);
    // online softmax update (per row; 16 column-lanes per row)
    float corr[8];
#pragma unroll
    for (int r = 0; r < 8; ++r) {
      float tmx = fmaxf(s0[r], s1[r]);
#pragma unroll
      for (int msk = 1; msk < 16; msk <<= 1) tmx = fmaxf(tmx, __shfl_xor(tmx, msk, 32));
      float mn = fmaxf(m[r], tmx);
      corr[r] = __expf(m[r] - mn);
      float p0 = __expf(s0[r] - mn), p1 = __expf(s1[r] - mn);
      float rs = p0 + p1;
#pragma unroll
      for (int msk = 1; msk < 16; msk <<= 1) rs += __shfl_xor(rs, msk, 32);
      l[r] = l[r] * corr[r] + rs;
      m[r] = mn;
      Pst[wid][r + off][lh]      = (__bf16)p0;   // C-layout -> LDS row-major
      Pst[wid][r + off][16 + lh] = (__bf16)p1;
    }
#pragma unroll
    for (int r = 0; r < 8; ++r) {
      O0[r] *= corr[r]; O1[r] *= corr[r]; O2[r] *= corr[r]; O3[r] *= corr[r];
    }
    asm volatile("s_wait_dscnt 0" ::: "memory");        // own-wave LDS RAW
    v16bf Pa = ld_a(&Pst[wid][lh][hs * 16], &Pst[wid][lh][hs * 16 + 8]);  // A-layout
    O0 = wmma_bf16(Pa, Bv0, O0);
    O1 = wmma_bf16(Pa, Bv1, O1);
    O2 = wmma_bf16(Pa, Bv2, O2);
    O3 = wmma_bf16(Pa, Bv3, O3);
  }
#pragma unroll
  for (int r = 0; r < 8; ++r) {
    float inv = 1.0f / l[r];
    int t = ibase + r + off;
    __bf16* dst = attnA + (size_t)t * D_MODEL + h * HDIM;
    dst[ 0 + lh] = (__bf16)(O0[r] * inv);
    dst[16 + lh] = (__bf16)(O1[r] * inv);
    dst[32 + lh] = (__bf16)(O2[r] * inv);
    dst[48 + lh] = (__bf16)(O3[r] * inv);
  }
}

// ---------------------------------------------------------------------------
// Kernel 6: output projection, attnA(2048x1024 bf16) @ wo^T -> fp32 d_out.
// Same 16x64-per-wave tiling + register double-buffering as k_proj.
// ---------------------------------------------------------------------------
__global__ void k_outproj(const __bf16* __restrict__ attnA,
                          const __bf16* __restrict__ wob, float* __restrict__ out) {
  int lane = threadIdx.x & 31, wid = threadIdx.x >> 5;
  int lh = lane & 15, hs = lane >> 4;
  int tile = blockIdx.x * 8 + wid;          // 2048 tiles = 128 x 16
  int tm = tile >> 4, tn = tile & 15;
  int row0 = tm * 16, col0 = tn * 64;
  const __bf16* arow = attnA + (size_t)(row0 + lh) * D_MODEL + hs * 16;
  const __bf16* br0 = wob + (size_t)(col0 +  0 + lh) * D_MODEL + hs * 16;
  const __bf16* br1 = wob + (size_t)(col0 + 16 + lh) * D_MODEL + hs * 16;
  const __bf16* br2 = wob + (size_t)(col0 + 32 + lh) * D_MODEL + hs * 16;
  const __bf16* br3 = wob + (size_t)(col0 + 48 + lh) * D_MODEL + hs * 16;
  v8f a0 = {}, a1 = {}, a2 = {}, a3 = {};
  v16bf ac  = ld_a(arow, arow + 8);
  v16bf b0c = ld_b(br0), b1c = ld_b(br1), b2c = ld_b(br2), b3c = ld_b(br3);
  for (int k = 32; k < D_MODEL; k += 32) {
    v16bf an  = ld_a(arow + k, arow + k + 8);
    v16bf b0n = ld_b(br0 + k), b1n = ld_b(br1 + k);
    v16bf b2n = ld_b(br2 + k), b3n = ld_b(br3 + k);
    a0 = wmma_bf16(ac, b0c, a0);
    a1 = wmma_bf16(ac, b1c, a1);
    a2 = wmma_bf16(ac, b2c, a2);
    a3 = wmma_bf16(ac, b3c, a3);
    ac = an; b0c = b0n; b1c = b1n; b2c = b2n; b3c = b3n;
  }
  a0 = wmma_bf16(ac, b0c, a0);
  a1 = wmma_bf16(ac, b1c, a1);
  a2 = wmma_bf16(ac, b2c, a2);
  a3 = wmma_bf16(ac, b3c, a3);
#pragma unroll
  for (int r = 0; r < 8; ++r) {
    size_t rowoff = (size_t)(row0 + r + hs * 8) * D_MODEL + col0;
    out[rowoff +  0 + lh] = a0[r];
    out[rowoff + 16 + lh] = a1[r];
    out[rowoff + 32 + lh] = a2[r];
    out[rowoff + 48 + lh] = a3[r];
  }
}

// ---------------------------------------------------------------------------
extern "C" void kernel_launch(void* const* d_in, const int* in_sizes, int n_in,
                              void* d_out, int out_size, void* d_ws, size_t ws_size,
                              hipStream_t stream) {
  const float* x  = (const float*)d_in[0];
  const float* U  = (const float*)d_in[1];
  const float* wq = (const float*)d_in[2];
  const float* wk = (const float*)d_in[3];
  const float* wv = (const float*)d_in[4];
  const float* wo = (const float*)d_in[5];
  float* out = (float*)d_out;

  char* ws = (char*)d_ws;
  size_t off = 0;
  auto alloc = [&](size_t bytes) -> void* {
    void* p = ws + off;
    off += (bytes + 255) & ~(size_t)255;
    return p;
  };
  __bf16* xb     = (__bf16*)alloc((size_t)T_SEQ * D_MODEL * 2);        // 4 MiB
  __bf16* wqb    = (__bf16*)alloc((size_t)D_MODEL * D_MODEL * 2);      // 2 MiB
  __bf16* wkb    = (__bf16*)alloc((size_t)D_MODEL * D_MODEL * 2);
  __bf16* wvb    = (__bf16*)alloc((size_t)D_MODEL * D_MODEL * 2);
  __bf16* wob    = (__bf16*)alloc((size_t)D_MODEL * D_MODEL * 2);
  __bf16* Qb     = (__bf16*)alloc((size_t)NHEAD * T_SEQ * HDIM * 2);   // 4 MiB
  __bf16* Kb     = (__bf16*)alloc((size_t)NHEAD * T_SEQ * HDIM * 2);
  __bf16* Vt     = (__bf16*)alloc((size_t)NHEAD * HDIM * T_SEQ * 2);
  __bf16* attnA  = (__bf16*)alloc((size_t)T_SEQ * D_MODEL * 2);
  float*  Sd     = (float*)alloc((size_t)NHEAD * 256 * 256 * 4);       // 4 MiB
  float*  base26 = (float*)alloc((size_t)NHEAD * 256 * 4);
  float*  us     = (float*)alloc((size_t)T_SEQ * 4);

  k_convert<<<(T_SEQ * D_MODEL + 255) / 256, 256, 0, stream>>>(
      x, U, wq, wk, wv, wo, xb, wqb, wkb, wvb, wob, us);
  k_proj<<<dim3(256, 3), 256, 0, stream>>>(xb, wqb, wkb, wvb, Qb, Kb, Vt);
  k_sd<<<dim3(32, NHEAD), 256, 0, stream>>>(Qb, Kb, Sd);
  k_kth<<<1024, 128, 0, stream>>>(Sd, base26);
  k_attn<<<512, 128, 0, stream>>>(Qb, Kb, Vt, Sd, base26, us, attnA);
  k_outproj<<<256, 256, 0, stream>>>(attnA, wob, out);
}
